// MultiHeadSelfAttention_17205638987928
// MI455X (gfx1250) — matrix operand order repaired and checked by reading, not compiled, not run on hardware
//
#include <hip/hip_runtime.h>

// ---------------------------------------------------------------------------
// MI455X (gfx1250, wave32) multi-head self-attention
//   stage 1: fp32 -> bf16 conversions (x, w_qkv, w_out)
//   stage 2: QKV GEMM  (v_wmma_f32_16x16x32_bf16), async LDS staging
//   stage 3: flash attention, TDM double-buffered K/V tiles, wmma QK^T / PV
//   stage 4: output projection GEMM + bias -> fp32 d_out
// ---------------------------------------------------------------------------

#define DIM    768
#define HEADS  12
#define DH     64
#define BATCH  2
#define SEQ    4096
#define QKVN   (3 * DIM)          // 2304
#define MTOT   (BATCH * SEQ)      // 8192

#define LOG2E  1.4426950408889634f
#define QSCALE 0.03608439182435161f   // 768^-0.5  (module scales by dim^-0.5)

typedef __attribute__((ext_vector_type(16))) __bf16 v16bf;
typedef __attribute__((ext_vector_type(8)))  float  v8f;

// gcc-style vectors: clang builtin params are vector_size vectors (distinct
// from ext_vector_type), so use these for all amdgcn builtin arguments.
typedef int          v4i_g  __attribute__((vector_size(16)));
typedef unsigned int v4u_g  __attribute__((vector_size(16)));
typedef int          v8i_g  __attribute__((vector_size(32)));

typedef v4i_g __attribute__((address_space(1)))* gptr128;  // global (non-const)
typedef v4i_g __attribute__((address_space(3)))* lptr128;  // LDS dst

#define HAS_ASYNC __has_builtin(__builtin_amdgcn_global_load_async_to_lds_b128)
#define HAS_TDM   __has_builtin(__builtin_amdgcn_tensor_load_to_lds)

static __device__ __forceinline__ v8f wmma_bf16(v16bf a, v16bf b, v8f c) {
  // D = A(16x32 bf16) * B(32x16 bf16) + C(16x16 f32)
  return __builtin_amdgcn_wmma_f32_16x16x32_bf16(false, a, false, b,
                                                 (short)0, c, false, false);
}

// low 32 bits of a generic LDS pointer == LDS byte offset (flat aperture rule)
static __device__ __forceinline__ unsigned lds_off(const void* p) {
  return (unsigned)(unsigned long long)p;
}
static __device__ __forceinline__ gptr128 as_g128(const void* p) {
  return (gptr128)(unsigned long long)p;
}
static __device__ __forceinline__ lptr128 as_l128(const void* p) {
  return (lptr128)(unsigned long long)lds_off(p);
}

static __device__ __forceinline__ void wait_asynccnt0() {
#if __has_builtin(__builtin_amdgcn_s_wait_asynccnt)
  __builtin_amdgcn_s_wait_asynccnt(0);
#else
  asm volatile("s_wait_asynccnt 0x0" ::: "memory");
#endif
}

#if HAS_TDM
// TDM descriptor load: one 64x64 bf16 tile of a [SEQ, 64] row-major tensor.
// D# built per CDNA5 ISA 8.3/8.4 (group0: count/lds/global/type, group1:
// data_size=2B, tensor_dim0=64, tensor_dim1=4096, tile 64x64, stride 64).
static __device__ __forceinline__ void tdm_load_64x64_bf16(unsigned lds_byte,
                                                           const __bf16* g) {
  unsigned long long ga = (unsigned long long)g;
  v4u_g g0 = {1u,                                    // count=1, user mode
              lds_byte,                              // lds_addr
              (unsigned)(ga & 0xffffffffu),          // global_addr[31:0]
              (unsigned)((ga >> 32) & 0x1ffffffu) |  // global_addr[56:32]
                  (2u << 30)};                       // type=2 ("image")
  v8i_g g1 = {(int)(1u << 16),                       // data_size=1 -> 2 bytes
              (int)(64u << 16),                      // tensor_dim0 = 64
              (int)(4096u << 16),                    // tensor_dim1 = 4096
              (int)(64u << 16),                      // tile_dim0 = 64
              64,                                    // tile_dim1 = 64
              64,                                    // tensor_dim0_stride = 64
              0, 0};
  v4i_g gz4 = {0, 0, 0, 0};
  v8i_g gz8 = {0, 0, 0, 0, 0, 0, 0, 0};
  __builtin_amdgcn_tensor_load_to_lds(g0, g1, gz4, gz4, gz8, 0);
}
#endif

// ---------------------------------------------------------------------------
// fp32 -> bf16 conversion (grid-stride)
// ---------------------------------------------------------------------------
__global__ void cvt_f32_bf16(const float* __restrict__ in,
                             __bf16* __restrict__ out, int n) {
  int i = blockIdx.x * blockDim.x + threadIdx.x;
  int stride = gridDim.x * blockDim.x;
  for (; i < n; i += stride) out[i] = (__bf16)in[i];
}

// ---------------------------------------------------------------------------
// Tiled bf16 GEMM: C[M,N] = A[M,K] @ B[K,N] (+bias epilogue)
// BM=128, BN=128, BK=32, 256 threads = 8 waves (2x4), each wave 64x32 tile
// MODE 0: out fp32 (+ bias)               -> output projection
// MODE 1: scatter q/k/v bf16 [b,h,n,64] (+bias, q *= QSCALE)
// ---------------------------------------------------------------------------
template <int MODE>
__global__ void __launch_bounds__(256)
gemm_bf16_kernel(const __bf16* __restrict__ A, const __bf16* __restrict__ B,
                 const float* __restrict__ bias, float* __restrict__ outf,
                 __bf16* __restrict__ qb, __bf16* __restrict__ kb,
                 __bf16* __restrict__ vb, int M, int N, int K) {
  __shared__ __align__(16) __bf16 sA[128 * 32];
  __shared__ __align__(16) __bf16 sB[32 * 128];

  const int tid  = threadIdx.x;
  const int lane = tid & 31;
  const int wv   = tid >> 5;
  const int hi   = lane >> 4;      // half-wave select
  const int ln   = lane & 15;
  const int wm   = wv >> 2;        // 0..1  (row group of 64)
  const int wn   = wv & 3;         // 0..3  (col group of 32)
  const int mb   = blockIdx.y;
  const int nb   = blockIdx.x;

  v8f acc[4][2];
  const v8f zacc = {0.f, 0.f, 0.f, 0.f, 0.f, 0.f, 0.f, 0.f};
#pragma unroll
  for (int i = 0; i < 4; i++)
#pragma unroll
    for (int j = 0; j < 2; j++) acc[i][j] = zacc;

  for (int k0 = 0; k0 < K; k0 += 32) {
    // ---- stage A tile 128x32 (8 KB) and B tile 32x128 (8 KB) into LDS ----
#if HAS_ASYNC
#pragma unroll
    for (int i = 0; i < 2; i++) {
      int c = tid + 256 * i;                 // 0..511 16B chunks
      int r = c >> 2, cc = c & 3;            // A: 4 chunks per row
      __builtin_amdgcn_global_load_async_to_lds_b128(
          as_g128(A + (size_t)(mb * 128 + r) * K + k0 + cc * 8),
          as_l128(sA + r * 32 + cc * 8), 0, 0);
    }
#pragma unroll
    for (int i = 0; i < 2; i++) {
      int c = tid + 256 * i;
      int r = c >> 4, cc = c & 15;           // B: 16 chunks per row
      __builtin_amdgcn_global_load_async_to_lds_b128(
          as_g128(B + (size_t)(k0 + r) * N + nb * 128 + cc * 8),
          as_l128(sB + r * 128 + cc * 8), 0, 0);
    }
    wait_asynccnt0();
#else
#pragma unroll
    for (int i = 0; i < 2; i++) {
      int c = tid + 256 * i;
      int r = c >> 2, cc = c & 3;
      *(uint4*)(sA + r * 32 + cc * 8) =
          *(const uint4*)(A + (size_t)(mb * 128 + r) * K + k0 + cc * 8);
    }
#pragma unroll
    for (int i = 0; i < 2; i++) {
      int c = tid + 256 * i;
      int r = c >> 4, cc = c & 15;
      *(uint4*)(sB + r * 128 + cc * 8) =
          *(const uint4*)(B + (size_t)(k0 + r) * N + nb * 128 + cc * 8);
    }
#endif
    __syncthreads();

    // ---- load fragments (ISA VGPR layouts) ----
    v16bf af[4];
#pragma unroll
    for (int mt = 0; mt < 4; mt++) {
      const __bf16* p = sA + (wm * 64 + mt * 16 + ln) * 32 + hi * 8;
#pragma unroll
      for (int e = 0; e < 8; e++) { af[mt][e] = p[e]; af[mt][e + 8] = p[e + 16]; }
    }
    v16bf bfg[2];
#pragma unroll
    for (int nt = 0; nt < 2; nt++) {
      int col = wn * 32 + nt * 16 + ln;
#pragma unroll
      for (int e = 0; e < 16; e++) bfg[nt][e] = sB[((e & 7) + 8 * hi + 16 * (e >> 3)) * 128 + col];
    }

#pragma unroll
    for (int mt = 0; mt < 4; mt++)
#pragma unroll
      for (int nt = 0; nt < 2; nt++)
        acc[mt][nt] = wmma_bf16(af[mt], bfg[nt], acc[mt][nt]);
    __syncthreads();
  }

  // ---- epilogue ----
#pragma unroll
  for (int mt = 0; mt < 4; mt++) {
#pragma unroll
    for (int nt = 0; nt < 2; nt++) {
#pragma unroll
      for (int r = 0; r < 8; r++) {
        int rg = mb * 128 + wm * 64 + mt * 16 + r + 8 * hi;  // C row = r + 8*hi
        int cg = nb * 128 + wn * 32 + nt * 16 + ln;          // C col = lane&15
        float val = acc[mt][nt][r] + bias[cg];
        if (MODE == 0) {
          outf[(size_t)rg * N + cg] = val;
        } else {
          int which = cg / DIM;            // 0=q 1=k 2=v
          int c7 = cg - which * DIM;
          int hh = c7 >> 6, dd = c7 & 63;
          int bi = rg >> 12, ni = rg & (SEQ - 1);
          size_t idx = ((size_t)(bi * HEADS + hh) * SEQ + ni) * DH + dd;
          if (which == 0)      qb[idx] = (__bf16)(val * QSCALE);
          else if (which == 1) kb[idx] = (__bf16)val;
          else                 vb[idx] = (__bf16)val;
        }
      }
    }
  }
}

// ---------------------------------------------------------------------------
// Flash attention. Grid: (SEQ/64, HEADS, BATCH), 128 threads = 4 waves.
// Each wave owns 16 query rows; block loops over 64-key tiles of K/V.
// K/V tiles are moved by the Tensor Data Mover with double buffering:
// wave 0 issues TDM loads for tile i+1, waits TENSORcnt<=2 (tile i done),
// barrier, all waves compute on tile i.
// ---------------------------------------------------------------------------
__global__ void __launch_bounds__(128)
flash_attn_kernel(const __bf16* __restrict__ Q, const __bf16* __restrict__ Kg,
                  const __bf16* __restrict__ Vg, __bf16* __restrict__ Ob) {
  __shared__ __align__(16) __bf16 sK[2 * 64 * 64];   // 16 KB (double buffered)
  __shared__ __align__(16) __bf16 sV[2 * 64 * 64];   // 16 KB
  __shared__ __align__(16) __bf16 sP[4 * 16 * 64];   // 8 KB per-wave P scratch

  const int tid  = threadIdx.x;
  const int lane = tid & 31;
  const int wv   = tid >> 5;
  const int hi   = lane >> 4;
  const int ln   = lane & 15;
  const int rb   = blockIdx.x;      // 64-row query block
  const int h    = blockIdx.y;
  const int bi   = blockIdx.z;

  const size_t head_off = (size_t)(bi * HEADS + h) * SEQ * DH;
  const __bf16* q = Q + head_off;
  const __bf16* k = Kg + head_off;
  const __bf16* v = Vg + head_off;

  const int row0 = rb * 64 + wv * 16;   // first query row of this wave

  // Q fragments stay in registers for the whole block (A layout)
  v16bf qf[2];
#pragma unroll
  for (int j = 0; j < 2; j++) {
    const __bf16* p = q + (size_t)(row0 + ln) * DH + 32 * j + 8 * hi;
#pragma unroll
    for (int e = 0; e < 8; e++) { qf[j][e] = p[e]; qf[j][e + 8] = p[e + 16]; }
  }

  const v8f zacc = {0.f, 0.f, 0.f, 0.f, 0.f, 0.f, 0.f, 0.f};
  v8f o[4];
#pragma unroll
  for (int t = 0; t < 4; t++) o[t] = zacc;
  float mrow[8], lrow[8];
#pragma unroll
  for (int r = 0; r < 8; r++) { mrow[r] = -1e30f; lrow[r] = 0.f; }

  const int NT = SEQ / 64;
#if HAS_TDM
  if (wv == 0) {                        // prologue: DMA tile 0 into buffer 0
    tdm_load_64x64_bf16(lds_off(sK), k);
    tdm_load_64x64_bf16(lds_off(sV), v);
  }
#endif

  for (int kbI = 0; kbI < NT; kbI++) {
    const int kb = kbI * 64;
    const int cur = kbI & 1;
    __bf16* sKc = sK + cur * (64 * 64);
    __bf16* sVc = sV + cur * (64 * 64);

#if HAS_TDM
    if (wv == 0) {
      if (kbI + 1 < NT) {               // DMA next tile into alt buffer
        tdm_load_64x64_bf16(lds_off(sK + (1 - cur) * (64 * 64)),
                            k + (size_t)(kb + 64) * DH);
        tdm_load_64x64_bf16(lds_off(sV + (1 - cur) * (64 * 64)),
                            v + (size_t)(kb + 64) * DH);
        __builtin_amdgcn_s_wait_tensorcnt(2);   // tile i landed, i+1 in flight
      } else {
        __builtin_amdgcn_s_wait_tensorcnt(0);
      }
    }
    __syncthreads();
#else
#pragma unroll
    for (int i = 0; i < 4; i++) {
      int c = tid + 128 * i;            // 0..511 chunks of 16B
      int r = c >> 3, cc = c & 7;
      *(uint4*)(sKc + r * 64 + cc * 8) =
          *(const uint4*)(k + (size_t)(kb + r) * DH + cc * 8);
      *(uint4*)(sVc + r * 64 + cc * 8) =
          *(const uint4*)(v + (size_t)(kb + r) * DH + cc * 8);
    }
    __syncthreads();
    if (kbI + 1 < NT) {                 // global_prefetch_b8 for next tile
      __builtin_prefetch(k + (size_t)(kb + 64 + (tid >> 1)) * DH + (tid & 1) * 32, 0, 0);
      __builtin_prefetch(v + (size_t)(kb + 64 + (tid >> 1)) * DH + (tid & 1) * 32, 0, 0);
    }
#endif

    // ---- S = Q @ K^T : 4 tiles of 16 keys, K-dim = 64 = 2 wmma steps ----
    v8f st[4];
#pragma unroll
    for (int t = 0; t < 4; t++) {
      st[t] = zacc;
#pragma unroll
      for (int j = 0; j < 2; j++) {
        v16bf kf;                        // B[K=d][N=key] = Ktile[key][d]
        const __bf16* p = sKc + (t * 16 + ln) * 64 + 32 * j + 8 * hi;
#pragma unroll
        for (int e = 0; e < 8; e++) { kf[e] = p[e]; kf[e + 8] = p[e + 16]; }
        st[t] = wmma_bf16(qf[j], kf, st[t]);
      }
    }

    // ---- online softmax (rows striped as r + 8*hi; reduce across 16 lanes) ----
    float alpha[8];
#pragma unroll
    for (int r = 0; r < 8; r++) {
      float mval = fmaxf(fmaxf(st[0][r], st[1][r]), fmaxf(st[2][r], st[3][r]));
#pragma unroll
      for (int msk = 1; msk < 16; msk <<= 1)
        mval = fmaxf(mval, __shfl_xor(mval, msk, 32));
      float nm = fmaxf(mrow[r], mval);
      alpha[r] = __builtin_amdgcn_exp2f((mrow[r] - nm) * LOG2E);
      mrow[r] = nm;
    }
    float rsum[8];
#pragma unroll
    for (int r = 0; r < 8; r++) rsum[r] = 0.f;
#pragma unroll
    for (int t = 0; t < 4; t++)
#pragma unroll
      for (int r = 0; r < 8; r++) {
        float pe = __builtin_amdgcn_exp2f((st[t][r] - mrow[r]) * LOG2E);
        st[t][r] = pe;
        rsum[r] += pe;
      }
#pragma unroll
    for (int r = 0; r < 8; r++) {
      float sv = rsum[r];
#pragma unroll
      for (int msk = 1; msk < 16; msk <<= 1) sv += __shfl_xor(sv, msk, 32);
      lrow[r] = lrow[r] * alpha[r] + sv;
    }
#pragma unroll
    for (int t = 0; t < 4; t++)
#pragma unroll
      for (int r = 0; r < 8; r++) o[t][r] *= alpha[r];

    // ---- P: C-layout -> A-layout via per-wave LDS scratch (bf16) ----
    __bf16* pw = sP + wv * (16 * 64);
#pragma unroll
    for (int t = 0; t < 4; t++)
#pragma unroll
      for (int r = 0; r < 8; r++)
        pw[(r + 8 * hi) * 64 + t * 16 + ln] = (__bf16)st[t][r];
    __builtin_amdgcn_wave_barrier();   // same-wave DS ops are in-order

    v16bf pf[2];
#pragma unroll
    for (int j = 0; j < 2; j++) {
      const __bf16* p = pw + ln * 64 + 32 * j + 8 * hi;
#pragma unroll
      for (int e = 0; e < 8; e++) { pf[j][e] = p[e]; pf[j][e + 8] = p[e + 16]; }
    }

    // ---- O += P @ V ----
#pragma unroll
    for (int t = 0; t < 4; t++)
#pragma unroll
      for (int j = 0; j < 2; j++) {
        v16bf vf;                        // B[K=key][N=d] = Vtile[key][d]
#pragma unroll
        for (int e = 0; e < 16; e++)
          vf[e] = sVc[(32 * j + (e & 7) + 8 * hi + 16 * (e >> 3)) * 64 + t * 16 + ln];
        o[t] = wmma_bf16(pf[j], vf, o[t]);
      }
    __syncthreads();
  }

  // ---- normalize, write bf16 concat-head activation for out-projection ----
#pragma unroll
  for (int r = 0; r < 8; r++) {
    float inv = 1.0f / lrow[r];
#pragma unroll
    for (int t = 0; t < 4; t++) {
      int row_g = bi * SEQ + rb * 64 + wv * 16 + r + 8 * hi;
      int col_g = h * DH + t * 16 + ln;
      Ob[(size_t)row_g * DIM + col_g] = (__bf16)(o[t][r] * inv);
    }
  }
}

// ---------------------------------------------------------------------------
// Host-side orchestration
// ---------------------------------------------------------------------------
extern "C" void kernel_launch(void* const* d_in, const int* in_sizes, int n_in,
                              void* d_out, int out_size, void* d_ws, size_t ws_size,
                              hipStream_t stream) {
  const float* x     = (const float*)d_in[0];   // [2,4096,768]
  const float* w_qkv = (const float*)d_in[1];   // [768,2304]
  const float* b_qkv = (const float*)d_in[2];   // [2304]
  const float* w_out = (const float*)d_in[3];   // [768,768]
  const float* b_out = (const float*)d_in[4];   // [768]
  float* out = (float*)d_out;                   // [2,4096,768] fp32

  char* ws = (char*)d_ws;
  size_t off = 0;
  auto take = [&](size_t bytes) -> void* {
    void* p = ws + off;
    off += (bytes + 255) & ~(size_t)255;
    return p;
  };
  __bf16* x_bf    = (__bf16*)take((size_t)MTOT * DIM * 2);
  __bf16* wqkv_bf = (__bf16*)take((size_t)DIM * QKVN * 2);
  __bf16* wout_bf = (__bf16*)take((size_t)DIM * DIM * 2);
  __bf16* q_bf    = (__bf16*)take((size_t)MTOT * DIM * 2);  // [b,h,n,64]
  __bf16* k_bf    = (__bf16*)take((size_t)MTOT * DIM * 2);
  __bf16* v_bf    = (__bf16*)take((size_t)MTOT * DIM * 2);
  __bf16* attn_bf = (__bf16*)take((size_t)MTOT * DIM * 2);  // [b*n,768]

  // stage 1: conversions
  cvt_f32_bf16<<<4096, 256, 0, stream>>>(x, x_bf, MTOT * DIM);
  cvt_f32_bf16<<<2048, 256, 0, stream>>>(w_qkv, wqkv_bf, DIM * QKVN);
  cvt_f32_bf16<<<1024, 256, 0, stream>>>(w_out, wout_bf, DIM * DIM);

  // stage 2: QKV projection, scatter into per-head layout (q pre-scaled)
  gemm_bf16_kernel<1><<<dim3(QKVN / 128, MTOT / 128), 256, 0, stream>>>(
      x_bf, wqkv_bf, b_qkv, nullptr, q_bf, k_bf, v_bf, MTOT, QKVN, DIM);

  // stage 3: flash attention (TDM double-buffered K/V)
  flash_attn_kernel<<<dim3(SEQ / 64, HEADS, BATCH), 128, 0, stream>>>(
      q_bf, k_bf, v_bf, attn_bf);

  // stage 4: output projection (fp32 out + bias)
  gemm_bf16_kernel<0><<<dim3(DIM / 128, MTOT / 128), 256, 0, stream>>>(
      attn_bf, wout_bf, b_out, out, nullptr, nullptr, nullptr, MTOT, DIM, DIM);
}
